// SwinBlock3D_50543175139544
// MI455X (gfx1250) — matrix operand order repaired and checked by reading, not compiled, not run on hardware
//
#include <hip/hip_runtime.h>
#include <hip/hip_bf16.h>

typedef __attribute__((ext_vector_type(16))) _Float16 v16h;
typedef __attribute__((ext_vector_type(8)))  float    v8f;

union HQ { v16h v; _Float16 h[16]; uint4 q[2]; };
union F8 { v8f  v; float    f[8];  };

#define TOK   98304     // 2 * 12 * 64 * 64 tokens
#define NWIN  2048      // 2 * 4 * 16 * 16 windows
#define WINN  48        // tokens per window
#define DIMC  192
#define QKVN  576
#define HIDN  768
#define SCALE 0.17677669529663687f   // 1/sqrt(32)

// ---------------------------------------------------------------------------
// region id for the shifted-window attention mask (computed analytically)
// t slices: [0,9) [9,11) [11,12) ; h/w slices: [0,60) [60,62) [62,64)
// ---------------------------------------------------------------------------
__device__ __forceinline__ int region_id(int it, int ih, int iw, int n) {
    int dt = n >> 4, dh = (n >> 2) & 3, dw = n & 3;
    int ts = it * 3 + dt, hs = ih * 4 + dh, ws = iw * 4 + dw;
    int rt = ts < 9 ? 0 : (ts < 11 ? 1 : 2);
    int rh = hs < 60 ? 0 : (hs < 62 ? 1 : 2);
    int rw = ws < 60 ? 0 : (ws < 62 ? 1 : 2);
    return rt * 9 + rh * 3 + rw;
}

// ---------------------------------------------------------------------------
// Pack fp32 weight [K,N] into per-lane WMMA B-fragment order (f16):
// out[((kb*(N/16) + ntile)*32 + lane)*16 + e]  = w[(kb*32 + grp*16 + e)*N + ntile*16 + (lane&15)]
// -> a B fragment is 32 contiguous bytes per lane, fully coalesced per wave.
// ---------------------------------------------------------------------------
__global__ void pack_w_kernel(const float* __restrict__ w, _Float16* __restrict__ out,
                              int K, int N) {
    int idx = blockIdx.x * 256 + threadIdx.x;
    if (idx >= K * N) return;
    int e    = idx & 15;
    int lane = (idx >> 4) & 31;
    int t    = idx >> 9;
    int ntiles = N >> 4;
    int ntile = t % ntiles;
    int kb    = t / ntiles;
    int n = ntile * 16 + (lane & 15);
    int k = kb * 32 + ((lane >> 4) & 1) * 16 + e;
    out[idx] = (_Float16)w[(size_t)k * N + n];
}

// zero the K-padding (keys 48..63) of the transposed V buffer [NWIN*6][32][64]
__global__ void vpad_zero_kernel(_Float16* __restrict__ Vt) {
    int idx = blockIdx.x * 256 + threadIdx.x;      // over 12288*32*16
    if (idx >= NWIN * 6 * 32 * 16) return;
    int key  = 48 + (idx & 15);
    int rest = idx >> 4;
    Vt[(size_t)rest * 64 + key] = (_Float16)0.f;
}

// ---------------------------------------------------------------------------
// transpose [b,c,t,h,w] -> shifted window tokens + LayerNorm1 -> f16
// ---------------------------------------------------------------------------
__global__ __launch_bounds__(256)
void prep_ln_kernel(const float* __restrict__ x, const float* __restrict__ g,
                    const float* __restrict__ bta, _Float16* __restrict__ Xs) {
    __shared__ float ld[DIMC * 64];
    __shared__ float mu_s[64], rs_s[64];
    int blk = blockIdx.x;               // 0..1535
    int bb  = blk / 768;
    int rem = blk % 768;
    int ts  = rem / 64;
    int hs  = rem % 64;
    int tsrc = ts + 1; if (tsrc >= 12) tsrc -= 12;
    int hsrc = (hs + 2) & 63;
    int tid  = threadIdx.x;

    for (int i = tid; i < DIMC * 64; i += 256) {
        int ch = i >> 6, wd = i & 63;
        int wsrc = (wd + 2) & 63;
        ld[ch * 64 + wd] =
            x[(((size_t)(bb * DIMC + ch) * 12 + tsrc) * 64 + hsrc) * 64 + wsrc];
    }
    __syncthreads();
    if (tid < 64) {
        float s = 0.f;
        for (int ch = 0; ch < DIMC; ++ch) s += ld[ch * 64 + tid];
        float mu = s * (1.f / DIMC);
        float v = 0.f;
        for (int ch = 0; ch < DIMC; ++ch) { float d = ld[ch * 64 + tid] - mu; v += d * d; }
        mu_s[tid] = mu;
        rs_s[tid] = rsqrtf(v * (1.f / DIMC) + 1e-5f);
    }
    __syncthreads();
    int it = ts / 3, dt = ts % 3, ih = hs >> 2, dh = hs & 3;
    for (int i = tid; i < DIMC * 64; i += 256) {
        int tok = i / DIMC, ch = i % DIMC;
        int iw = tok >> 2, dw = tok & 3;
        int win = ((bb * 4 + it) * 16 + ih) * 16 + iw;
        int n   = dt * 16 + dh * 4 + dw;
        float v = (ld[ch * 64 + tok] - mu_s[tok]) * rs_s[tok] * g[ch] + bta[ch];
        Xs[((size_t)(win * WINN + n)) * DIMC + ch] = (_Float16)v;
    }
}

// ---------------------------------------------------------------------------
// LayerNorm2 fp32 -> f16, one wave per token
// ---------------------------------------------------------------------------
__global__ __launch_bounds__(256)
void ln2_kernel(const float* __restrict__ xin, const float* __restrict__ g,
                const float* __restrict__ bta, _Float16* __restrict__ out) {
    int wv = threadIdx.x >> 5, lane = threadIdx.x & 31;
    size_t tok = (size_t)blockIdx.x * 8 + wv;
    const float* row = xin + tok * DIMC;
    float v[6]; float s = 0.f;
#pragma unroll
    for (int j = 0; j < 6; ++j) { v[j] = row[lane + 32 * j]; s += v[j]; }
#pragma unroll
    for (int o = 16; o >= 1; o >>= 1) s += __shfl_xor(s, o, 32);
    float mu = s * (1.f / DIMC);
    float vr = 0.f;
#pragma unroll
    for (int j = 0; j < 6; ++j) { float d = v[j] - mu; vr += d * d; }
#pragma unroll
    for (int o = 16; o >= 1; o >>= 1) vr += __shfl_xor(vr, o, 32);
    float rstd = rsqrtf(vr * (1.f / DIMC) + 1e-5f);
    _Float16* orow = out + tok * DIMC;
#pragma unroll
    for (int j = 0; j < 6; ++j) {
        int ch = lane + 32 * j;
        orow[ch] = (_Float16)((v[j] - mu) * rstd * g[ch] + bta[ch]);
    }
}

// ---------------------------------------------------------------------------
// WMMA GEMM, no LDS / no barriers: everything is L2-resident (192MB L2).
// block = 256 thr = 8 waves; block tile 128(M) x 64(N); wave tile 32x32
// A[M,K] f16 row-major (fragment-loaded directly, b128s), Bp pre-packed.
// ---------------------------------------------------------------------------
#define EPI_QKV  0
#define EPI_GELU 1
#define EPI_PROJ 2
#define EPI_FC2  3

template <int EPI, int K, int N>
__global__ __launch_bounds__(256)
void gemm_wmma_kernel(const _Float16* __restrict__ A, const _Float16* __restrict__ Bp,
                      const float* __restrict__ bias, _Float16* __restrict__ outH,
                      float* __restrict__ outF, const float* __restrict__ xorig,
                      const float* __restrict__ resid, float* __restrict__ dout,
                      _Float16* __restrict__ vtOut) {
    int tid  = threadIdx.x;
    int wv   = tid >> 5, lane = tid & 31;
    int wm   = wv & 3, wn = wv >> 2;
    int mbase = blockIdx.x * 128 + wm * 32;
    int nbase = blockIdx.y * 64 + wn * 32;
    int grp   = (lane >> 4) & 1;

    F8 acc[2][2];
#pragma unroll
    for (int mt = 0; mt < 2; ++mt)
#pragma unroll
        for (int nt = 0; nt < 2; ++nt)
#pragma unroll
            for (int r = 0; r < 8; ++r) acc[mt][nt].f[r] = 0.f;

    const int NT16 = N >> 4;
    int ntile0 = nbase >> 4;

    for (int kb = 0; kb < (K >> 5); ++kb) {
        int kc = kb << 5;
        HQ a[2], b[2];
#pragma unroll
        for (int mt = 0; mt < 2; ++mt) {
            const _Float16* p =
                A + (size_t)(mbase + mt * 16 + (lane & 15)) * K + kc + grp * 16;
            a[mt].q[0] = *(const uint4*)p;
            a[mt].q[1] = *(const uint4*)(p + 8);
        }
#pragma unroll
        for (int nt = 0; nt < 2; ++nt) {
            const _Float16* p =
                Bp + ((size_t)(kb * NT16 + ntile0 + nt) * 32 + lane) * 16;
            b[nt].q[0] = *(const uint4*)p;
            b[nt].q[1] = *(const uint4*)(p + 8);
        }
#pragma unroll
        for (int mt = 0; mt < 2; ++mt)
#pragma unroll
            for (int nt = 0; nt < 2; ++nt)
                acc[mt][nt].v = __builtin_amdgcn_wmma_f32_16x16x32_f16(
                    false, a[mt].v, false, b[nt].v, (short)0, acc[mt][nt].v,
                    false, false);
    }

#pragma unroll
    for (int mt = 0; mt < 2; ++mt)
#pragma unroll
        for (int nt = 0; nt < 2; ++nt)
#pragma unroll
            for (int r = 0; r < 8; ++r) {
                int nloc = lane & 15, mloc = grp * 8 + r;
                int gm = mbase + mt * 16 + mloc;
                int gn = nbase + nt * 16 + nloc;
                float val = acc[mt][nt].f[r] + bias[gn];
                if (EPI == EPI_QKV) {
                    if (gn < 2 * DIMC) {
                        outH[(size_t)gm * N + gn] = (_Float16)val;
                    } else {
                        // store V transposed + K-padded: Vt[win*6+hd][d][64]
                        int win = gm / WINN, key = gm % WINN;
                        int hd = (gn - 2 * DIMC) >> 5;
                        int d  = (gn - 2 * DIMC) & 31;
                        vtOut[((size_t)(win * 6 + hd) * 32 + d) * 64 + key] =
                            (_Float16)val;
                    }
                } else if (EPI == EPI_GELU) {
                    val = 0.5f * val * (1.f + erff(val * 0.70710678118654752f));
                    outH[(size_t)gm * N + gn] = (_Float16)val;
                } else if (EPI == EPI_PROJ) {
                    // window token -> reverse partition + roll(+1,+2,+2),
                    // add shortcut from channel-first x, write fp32 token-major.
                    int win = gm / WINN, n = gm % WINN;
                    int bb  = win >> 10;
                    int wib = win & 1023;
                    int it = wib >> 8, ih = (wib >> 4) & 15, iw = wib & 15;
                    int dt = n >> 4, dh = (n >> 2) & 3, dw = n & 3;
                    int ts = it * 3 + dt, hs = ih * 4 + dh, wsc = iw * 4 + dw;
                    int t = ts + 1; if (t >= 12) t -= 12;
                    int h = (hs + 2) & 63;
                    int w = (wsc + 2) & 63;
                    float sc =
                        xorig[(((size_t)(bb * DIMC + gn) * 12 + t) * 64 + h) * 64 + w];
                    int tok = ((bb * 12 + t) * 64 + h) * 64 + w;
                    outF[(size_t)tok * DIMC + gn] = val + sc;
                } else { // EPI_FC2: residual + transpose to [b,c,t,h,w]
                    float res = resid[(size_t)gm * DIMC + gn];
                    int bb = gm / 49152;
                    int r2 = gm % 49152;
                    int t  = r2 / 4096;
                    int hw = r2 % 4096;
                    dout[((size_t)(bb * DIMC + gn) * 12 + t) * 4096 + hw] = val + res;
                }
            }
}

// ---------------------------------------------------------------------------
// Attention: one block per window (192 threads), one wave per head.
// S = QK^T*scale + mask ; softmax ; O = P V
// ---------------------------------------------------------------------------
__global__ __launch_bounds__(192)
void attn_kernel(const _Float16* __restrict__ QKV, const _Float16* __restrict__ Vt,
                 _Float16* __restrict__ Aout) {
    __shared__ float S[6][WINN][WINN];     // 54 KB
    int hd = threadIdx.x >> 5, lane = threadIdx.x & 31;
    int win = blockIdx.x;
    int wib = win & 1023;
    int it = wib >> 8, ih = (wib >> 4) & 15, iw = wib & 15;
    size_t base = (size_t)win * WINN;

    H16_FRAGS:;
    HQ Aq[3], Bk[3];
#pragma unroll
    for (int mt = 0; mt < 3; ++mt)
#pragma unroll
        for (int e = 0; e < 16; ++e) {
            int j = e >> 1;
            int m = (lane & 15) + mt * 16;
            int k = ((lane >> 4) & 1) * 16 + e;
            Aq[mt].h[e] = QKV[(base + m) * QKVN + hd * 32 + k];
        }
#pragma unroll
    for (int nt = 0; nt < 3; ++nt)
#pragma unroll
        for (int e = 0; e < 16; ++e) {
            int n = (lane & 15) + nt * 16;
            int k = ((lane >> 4) & 1) * 16 + e;
            Bk[nt].h[e] = QKV[(base + n) * QKVN + DIMC + hd * 32 + k];
        }

#pragma unroll
    for (int mt = 0; mt < 3; ++mt)
#pragma unroll
        for (int nt = 0; nt < 3; ++nt) {
            F8 acc;
#pragma unroll
            for (int r = 0; r < 8; ++r) acc.f[r] = 0.f;
            acc.v = __builtin_amdgcn_wmma_f32_16x16x32_f16(
                false, Aq[mt].v, false, Bk[nt].v, (short)0, acc.v, false, false);
#pragma unroll
            for (int r = 0; r < 8; ++r) {
                int nloc = lane & 15, mloc = ((lane >> 4) & 1) * 8 + r;
                int rtok = mt * 16 + mloc, ctok = nt * 16 + nloc;
                float mv = (region_id(it, ih, iw, rtok) == region_id(it, ih, iw, ctok))
                               ? 0.f : -100.f;
                S[hd][rtok][ctok] = acc.f[r] * SCALE + mv;
            }
        }
    __syncthreads();

    // softmax per row (over 48 keys)
    for (int row = lane; row < WINN; row += 32) {
        float mx = -1e30f;
        for (int c = 0; c < WINN; ++c) mx = fmaxf(mx, S[hd][row][c]);
        float sm = 0.f;
        for (int c = 0; c < WINN; ++c) {
            float e = __expf(S[hd][row][c] - mx);
            S[hd][row][c] = e;
            sm += e;
        }
        float inv = 1.f / sm;
        for (int c = 0; c < WINN; ++c) S[hd][row][c] *= inv;
    }
    __syncthreads();

    // O = P @ V ; K-dim 48 padded to 64 ; V pre-transposed+padded -> b128 loads
#pragma unroll
    for (int mt = 0; mt < 3; ++mt)
#pragma unroll
        for (int dt2 = 0; dt2 < 2; ++dt2) {
            F8 acc;
#pragma unroll
            for (int r = 0; r < 8; ++r) acc.f[r] = 0.f;
#pragma unroll
            for (int ks = 0; ks < 2; ++ks) {
                HQ a, b;
#pragma unroll
                for (int e = 0; e < 16; ++e) {
                    int j = e >> 1;
                    int m = (lane & 15) + mt * 16;
                    int k = ks * 32 + ((lane >> 4) & 1) * 16 + e;
                    a.h[e] = (k < WINN) ? (_Float16)S[hd][m][k] : (_Float16)0.f;
                }
#pragma unroll
                for (int e = 0; e < 16; ++e) {
                    int n = (lane & 15) + dt2 * 16;
                    int k = ks * 32 + ((lane >> 4) & 1) * 16 + e;
                    b.h[e] = Vt[((size_t)(win * 6 + hd) * 32 + n) * 64 + k];
                }
                acc.v = __builtin_amdgcn_wmma_f32_16x16x32_f16(
                    false, a.v, false, b.v, (short)0, acc.v, false, false);
            }
#pragma unroll
            for (int r = 0; r < 8; ++r) {
                int nloc = lane & 15, mloc = ((lane >> 4) & 1) * 8 + r;
                int tokr = mt * 16 + mloc, d = dt2 * 16 + nloc;
                Aout[(base + tokr) * DIMC + hd * 32 + d] = (_Float16)acc.f[r];
            }
        }
}

// ---------------------------------------------------------------------------
// host side
// ---------------------------------------------------------------------------
extern "C" void kernel_launch(void* const* d_in, const int* in_sizes, int n_in,
                              void* d_out, int out_size, void* d_ws, size_t ws_size,
                              hipStream_t stream) {
    const float* x       = (const float*)d_in[0];
    const float* norm1_g = (const float*)d_in[1];
    const float* norm1_b = (const float*)d_in[2];
    const float* qkv_w   = (const float*)d_in[3];
    const float* qkv_b   = (const float*)d_in[4];
    const float* proj_w  = (const float*)d_in[5];
    const float* proj_b  = (const float*)d_in[6];
    const float* norm2_g = (const float*)d_in[7];
    const float* norm2_b = (const float*)d_in[8];
    const float* fc1_w   = (const float*)d_in[9];
    const float* fc1_b   = (const float*)d_in[10];
    const float* fc2_w   = (const float*)d_in[11];
    const float* fc2_b   = (const float*)d_in[12];

    char* ws = (char*)d_ws;
    size_t offA  = 0;                                   // f16 [TOK,768] (QK out, then h_mid)
    size_t offB  = offA  + (size_t)TOK * HIDN * 2;      // f16 [TOK,192] (Xs, attn-out, LN2-out)
    size_t offC  = offB  + (size_t)TOK * DIMC * 2;      // f32 [TOK,192] (post-attn residual)
    size_t offVt = offC  + (size_t)TOK * DIMC * 4;      // f16 [NWIN*6][32][64] transposed V
    size_t offWq = offVt + (size_t)NWIN * 6 * 32 * 64 * 2;
    size_t offWp = offWq + (size_t)DIMC * QKVN * 2;
    size_t offW1 = offWp + (size_t)DIMC * DIMC * 2;
    size_t offW2 = offW1 + (size_t)DIMC * HIDN * 2;

    _Float16* bufA = (_Float16*)(ws + offA);
    _Float16* bufB = (_Float16*)(ws + offB);
    float*    bufC = (float*)   (ws + offC);
    _Float16* Vt   = (_Float16*)(ws + offVt);
    _Float16* Wq16 = (_Float16*)(ws + offWq);
    _Float16* Wp16 = (_Float16*)(ws + offWp);
    _Float16* W116 = (_Float16*)(ws + offW1);
    _Float16* W216 = (_Float16*)(ws + offW2);

    // pack weights into WMMA B-fragment order (f16)
    pack_w_kernel<<<(DIMC * QKVN + 255) / 256, 256, 0, stream>>>(qkv_w, Wq16, DIMC, QKVN);
    pack_w_kernel<<<(DIMC * DIMC + 255) / 256, 256, 0, stream>>>(proj_w, Wp16, DIMC, DIMC);
    pack_w_kernel<<<(DIMC * HIDN + 255) / 256, 256, 0, stream>>>(fc1_w, W116, DIMC, HIDN);
    pack_w_kernel<<<(HIDN * DIMC + 255) / 256, 256, 0, stream>>>(fc2_w, W216, HIDN, DIMC);

    // zero padded keys of Vt (deterministic each call)
    vpad_zero_kernel<<<(NWIN * 6 * 32 * 16 + 255) / 256, 256, 0, stream>>>(Vt);

    // shift + window partition + LN1 -> f16 window tokens
    prep_ln_kernel<<<1536, 256, 0, stream>>>(x, norm1_g, norm1_b, bufB);

    // QKV projection (V stored transposed+padded into Vt)
    gemm_wmma_kernel<EPI_QKV, DIMC, QKVN><<<dim3(TOK / 128, QKVN / 64), 256, 0, stream>>>(
        bufB, Wq16, qkv_b, bufA, nullptr, nullptr, nullptr, nullptr, Vt);

    // windowed attention
    attn_kernel<<<NWIN, 192, 0, stream>>>(bufA, Vt, bufB);

    // output projection + reverse shift + residual (fp32, spatial token order)
    gemm_wmma_kernel<EPI_PROJ, DIMC, DIMC><<<dim3(TOK / 128, DIMC / 64), 256, 0, stream>>>(
        bufB, Wp16, proj_b, nullptr, bufC, x, nullptr, nullptr, nullptr);

    // LN2 -> f16
    ln2_kernel<<<TOK / 8, 256, 0, stream>>>(bufC, norm2_g, norm2_b, bufB);

    // fc1 + exact GELU
    gemm_wmma_kernel<EPI_GELU, DIMC, HIDN><<<dim3(TOK / 128, HIDN / 64), 256, 0, stream>>>(
        bufB, W116, fc1_b, bufA, nullptr, nullptr, nullptr, nullptr, nullptr);

    // fc2 + residual + transpose to [b,c,t,h,w] output
    gemm_wmma_kernel<EPI_FC2, HIDN, DIMC><<<dim3(TOK / 128, DIMC / 64), 256, 0, stream>>>(
        bufA, W216, fc2_b, nullptr, nullptr, nullptr, bufC, (float*)d_out, nullptr);

    (void)in_sizes; (void)n_in; (void)out_size; (void)ws_size;
}
